// Attention_55765855372115
// MI455X (gfx1250) — matrix operand order repaired and checked by reading, not compiled, not run on hardware
//
#include <hip/hip_runtime.h>
#include <hip/hip_bf16.h>

typedef __bf16 bf16_t;
typedef __attribute__((ext_vector_type(16))) __bf16 v16bf;
typedef __attribute__((ext_vector_type(8)))  __bf16 v8bf;
typedef __attribute__((ext_vector_type(8)))  float  v8f;
typedef __attribute__((ext_vector_type(4)))  unsigned int v4u;

#define B_   32
#define C_   256
#define N_   1024
#define G_   8
#define CG_  32
#define OC3_ 768
#define SCALE_ 0.0625f   /* 1/sqrt(256) */
#define EPS_ 1e-5f

// ---------------------------------------------------------------------------
// CDNA5 async LDS staging (GLOBAL_LOAD_ASYNC_TO_LDS_B128, ASYNCcnt-tracked).
// Each lane copies 16B from its global address to its LDS offset.
// ---------------------------------------------------------------------------
__device__ __forceinline__ void async_b128(unsigned lds_off, const void* gaddr) {
  asm volatile("global_load_async_to_lds_b128 %0, %1, off"
               :: "v"(lds_off), "v"(gaddr) : "memory");
}
__device__ __forceinline__ unsigned lds_addr_of(const void* p) {
  // generic LDS pointer carries the LDS byte offset in bits [31:0]
  return (unsigned)(uintptr_t)p;
}

// ---------------------------------------------------------------------------
// WMMA fragment loaders (wave32; A/B use this code's contiguous k order, same in A and B; C/D as in the ISA)
// A (16x32 bf16): lane m = lane&15; k(e) = e + hi*16 (contiguous per lane)
// B (32x16 bf16): lane n = lane&15; k(e) = e + hi*16 (contiguous per lane)
// C/D (16x16 f32): lane n = lane&15; vgpr j -> row m = j + hi*8
// ---------------------------------------------------------------------------
__device__ __forceinline__ v16bf load_a_bf16(const bf16_t* __restrict__ base,
                                             int lda, int lane) {
  const int m  = lane & 15;
  const int hi = lane >> 4;
  const bf16_t* p = base + (size_t)m * lda + hi * 16;
  v8bf lo = *(const v8bf*)(p);
  v8bf hh = *(const v8bf*)(p + 8);
  v16bf a;
#pragma unroll
  for (int e = 0; e < 8; ++e) { a[e] = lo[e]; a[8 + e] = hh[e]; }
  return a;
}

__device__ __forceinline__ v16bf load_b_ctg(const bf16_t* __restrict__ base,
                                            int stride, int lane) {
  const int n  = lane & 15;
  const int hi = lane >> 4;
  const bf16_t* p = base + (size_t)n * stride + hi * 16;
  v8bf lo = *(const v8bf*)(p);
  v8bf hh = *(const v8bf*)(p + 8);
  v16bf b;
#pragma unroll
  for (int e = 0; e < 8; ++e) { b[e] = lo[e]; b[8 + e] = hh[e]; }
  return b;
}

// ---------------------------------------------------------------------------
// Kernel 0: one-time f32 -> bf16 weight conversion (keeps cvt out of GEMMs).
// ---------------------------------------------------------------------------
__global__ void cvt_kernel(const float* __restrict__ src,
                           bf16_t* __restrict__ dst, int n) {
  int i = blockIdx.x * 256 + threadIdx.x;
  if (i < n) dst[i] = (bf16_t)src[i];
}

// ---------------------------------------------------------------------------
// Kernel 1: GroupNorm. One block per (batch, group). Writes xn f32 [c][n] for
// the residual path and xnT bf16 [n][c] for WMMA B operands.
// ---------------------------------------------------------------------------
__global__ void gn_kernel(const float* __restrict__ x,
                          const float* __restrict__ gw,
                          const float* __restrict__ gb,
                          float* __restrict__ xn,
                          bf16_t* __restrict__ xnT) {
  const int bg = blockIdx.x;             // 0..255
  const int b = bg / G_, g = bg % G_;
  const int tid = threadIdx.x;           // 256 threads
  const float* xp = x + ((size_t)b * C_ + (size_t)g * CG_) * N_;

  float s = 0.f, ss = 0.f;
  for (int i = tid; i < CG_ * N_; i += 256) {
    float v = xp[i];
    s += v; ss += v * v;
  }
  __shared__ float rs[256], rss[256];
  rs[tid] = s; rss[tid] = ss;
  __syncthreads();
  for (int off = 128; off > 0; off >>= 1) {
    if (tid < off) { rs[tid] += rs[tid + off]; rss[tid] += rss[tid + off]; }
    __syncthreads();
  }
  const float inv_n = 1.0f / (float)(CG_ * N_);
  const float mean = rs[0] * inv_n;
  const float var  = rss[0] * inv_n - mean * mean;
  const float rstd = rsqrtf(var + EPS_);

  float* xnp = xn + ((size_t)b * C_ + (size_t)g * CG_) * N_;
  bf16_t* xtp = xnT + (size_t)b * N_ * C_;
  for (int i = tid; i < CG_ * N_; i += 256) {
    int cc = i >> 10;            // i / N_
    int n  = i & (N_ - 1);
    int c  = g * CG_ + cc;
    float v = (xp[i] - mean) * rstd * gw[c] + gb[c];
    xnp[i] = v;
    xtp[(size_t)n * C_ + c] = (bf16_t)v;
  }
}

// ---------------------------------------------------------------------------
// Kernel 2: QKV projection. D[768,1024] = W[768,256] x Xn[256,1024] + bias,
// per batch. Register-blocked: one 32x64 tile per wave (2x4 WMMA tiles,
// 8 f32 accumulators = 64 VGPRs -> no spills), K=256 fully unrolled.
// Q gets 1/sqrt(c) folded in. Q,K stored bf16 [n][c]; V stored bf16 [c][n]
// (pre-transposed so the attention V tile is an element-contiguous copy).
// ---------------------------------------------------------------------------
__global__ void qkv_kernel(const bf16_t* __restrict__ xnT,
                           const bf16_t* __restrict__ qw_bf,
                           const float* __restrict__ qbias,
                           bf16_t* __restrict__ qo,
                           bf16_t* __restrict__ ko,
                           bf16_t* __restrict__ vT) {
  const int wave = (blockIdx.x * blockDim.x + threadIdx.x) >> 5;  // 0..12287
  const int lane = threadIdx.x & 31;
  const int nt4 = wave & 15;            // 16 x 64-col strips
  const int t2  = wave >> 4;
  const int mt2 = t2 % 24;              // 24 x 32-row strips (768 rows)
  const int b   = t2 / 24;

  const bf16_t* Bb = xnT + (size_t)b * N_ * C_ + (size_t)(nt4 * 64) * C_;
  const bf16_t* Ab = qw_bf + (size_t)(mt2 * 32) * C_;

  v8f Acc[2][4];
#pragma unroll
  for (int i = 0; i < 2; ++i)
#pragma unroll
    for (int j = 0; j < 4; ++j) Acc[i][j] = (v8f){};

#pragma unroll
  for (int kk = 0; kk < C_; kk += 32) {
    v16bf Af[2], Bf[4];
#pragma unroll
    for (int i = 0; i < 2; ++i)
      Af[i] = load_a_bf16(Ab + (size_t)(i * 16) * C_ + kk, C_, lane);
#pragma unroll
    for (int j = 0; j < 4; ++j)
      Bf[j] = load_b_ctg(Bb + (size_t)(j * 16) * C_ + kk, C_, lane);
#pragma unroll
    for (int i = 0; i < 2; ++i)
#pragma unroll
      for (int j = 0; j < 4; ++j)
        Acc[i][j] = __builtin_amdgcn_wmma_f32_16x16x32_bf16(
            false, Af[i], false, Bf[j], (short)0, Acc[i][j], false, false);
  }

  const int hi = lane >> 4, ln = lane & 15;
#pragma unroll
  for (int i = 0; i < 2; ++i) {
    const int orow0 = mt2 * 32 + i * 16 + hi * 8;  // 8 consecutive channels
    const int seg = orow0 >> 8;                    // 0=q 1=k 2=v
    const int ch0 = orow0 & 255;
    const float scl = (seg == 0) ? SCALE_ : 1.0f;
#pragma unroll
    for (int j = 0; j < 4; ++j) {
      const int ncol = nt4 * 64 + j * 16 + ln;
      if (seg < 2) {
        bf16_t* dst = (seg == 0 ? qo : ko) +
                      (size_t)b * N_ * C_ + (size_t)ncol * C_ + ch0;
        v8bf outv;
#pragma unroll
        for (int r = 0; r < 8; ++r)
          outv[r] = (bf16_t)((Acc[i][j][r] + qbias[orow0 + r]) * scl);
        *(v8bf*)dst = outv;
      } else {
        bf16_t* dst = vT + (size_t)b * C_ * N_ + (size_t)ch0 * N_ + ncol;
#pragma unroll
        for (int r = 0; r < 8; ++r)
          dst[(size_t)r * N_] = (bf16_t)(Acc[i][j][r] + qbias[orow0 + r]);
      }
    }
  }
}

// ---------------------------------------------------------------------------
// Kernel 3: Flash attention with async double-buffered K/V staging.
// Block = 1 batch x 128 q-rows; 8 waves, each owning a 16-row q tile and
// 128 accumulator VGPRs. Per 32-key step: next K tile (16KB, contiguous) and
// next V^T tile (256 rows x 64B) are staged via global_load_async_to_lds_b128
// while the current tile feeds 32 WMMAs + online softmax.
// ---------------------------------------------------------------------------
__global__ void __launch_bounds__(256, 1)
attn_kernel(const bf16_t* __restrict__ q,
            const bf16_t* __restrict__ k,
            const bf16_t* __restrict__ vT,
            bf16_t* __restrict__ o) {
  __shared__ bf16_t Kt[2][32 * 256];   // [buf][keyLocal][c]   32 KB
  __shared__ bf16_t Vt[2][256 * 32];   // [buf][c][keyLocal]   32 KB
  __shared__ bf16_t Pt[8][16 * 32];    // per-wave P tile       8 KB

  const int blk = blockIdx.x;          // 256 blocks
  const int b  = blk >> 3;
  const int qb = blk & 7;
  const int tid = threadIdx.x;
  const int wv = tid >> 5, lane = tid & 31;
  const int hi = lane >> 4, ln = lane & 15;
  const int qrow0 = qb * 128 + wv * 16;

  const bf16_t* qp  = q  + (size_t)b * N_ * C_ + (size_t)qrow0 * C_;
  const bf16_t* kb  = k  + (size_t)b * N_ * C_;
  const bf16_t* vTb = vT + (size_t)b * C_ * N_;

  // Persistent Q A-fragments: 8 x (16x32) covering K=256
  v16bf Qf[8];
#pragma unroll
  for (int ck = 0; ck < 8; ++ck)
    Qf[ck] = load_a_bf16(qp + ck * 32, C_, lane);

  v8f O[16];
#pragma unroll
  for (int i = 0; i < 16; ++i) O[i] = (v8f){};
  float mrow[8], lrow[8];
#pragma unroll
  for (int j = 0; j < 8; ++j) { mrow[j] = -1e30f; lrow[j] = 0.f; }

  // --- async staging of one 32-key stage (8 b128 asyncs per thread) -------
  auto stage = [&](int kp, int buf) {
    // K tile: keys [kp*32, +32) of k[b][n][c] -> one contiguous 16KB block
    {
      const char* g = (const char*)(kb + (size_t)kp * 32 * C_);
      unsigned l = lds_addr_of(&Kt[buf][0]);
#pragma unroll
      for (int j = 0; j < 4; ++j) {
        int idx = tid * 4 + j;                       // 0..1023 x 16B
        async_b128(l + idx * 16, g + (size_t)idx * 16);
      }
    }
    // V^T tile: row c=tid of vT[b][c][n], 64 contiguous bytes
    {
      const char* g = (const char*)(vTb + (size_t)tid * N_ + kp * 32);
      unsigned l = lds_addr_of(&Vt[buf][0]) + tid * 64;
#pragma unroll
      for (int j = 0; j < 4; ++j)
        async_b128(l + j * 16, g + (size_t)j * 16);
    }
  };

  stage(0, 0);

  for (int kp = 0; kp < 32; ++kp) {     // 32 keys per step
    const int cur = kp & 1;
    if (kp + 1 < 32) {
      stage(kp + 1, 1 - cur);
      asm volatile("s_wait_asynccnt 0x8" ::: "memory");  // current buf landed
    } else {
      asm volatile("s_wait_asynccnt 0x0" ::: "memory");
    }
    __syncthreads();

    // --- scores: two 16x16 tiles over 32 keys --------------------------
    v8f S[2];
#pragma unroll
    for (int t = 0; t < 2; ++t) {
      v8f s = {};
#pragma unroll
      for (int ck = 0; ck < 8; ++ck) {
        const bf16_t* p = &Kt[cur][(size_t)(t * 16 + ln) * C_ + ck * 32 + hi * 16];
        v8bf lo = *(const v8bf*)(p);
        v8bf hh = *(const v8bf*)(p + 8);
        v16bf bb;
#pragma unroll
        for (int e = 0; e < 8; ++e) { bb[e] = lo[e]; bb[8 + e] = hh[e]; }
        s = __builtin_amdgcn_wmma_f32_16x16x32_bf16(false, Qf[ck], false, bb,
                                                    (short)0, s, false, false);
      }
      S[t] = s;
    }

    // --- online softmax (row j+hi*8 lives in vgpr j of this half) ------
    float alpha[8];
#pragma unroll
    for (int j = 0; j < 8; ++j) {
      float mx = fmaxf(S[0][j], S[1][j]);
#pragma unroll
      for (int msk = 1; msk < 16; msk <<= 1)
        mx = fmaxf(mx, __shfl_xor(mx, msk, 32));
      float nm = fmaxf(mrow[j], mx);
      alpha[j] = __expf(mrow[j] - nm);
      float p0 = __expf(S[0][j] - nm);
      float p1 = __expf(S[1][j] - nm);
      float rs = p0 + p1;
#pragma unroll
      for (int msk = 1; msk < 16; msk <<= 1)
        rs += __shfl_xor(rs, msk, 32);
      lrow[j] = lrow[j] * alpha[j] + rs;
      mrow[j] = nm;
      S[0][j] = p0; S[1][j] = p1;
    }
#pragma unroll
    for (int ct = 0; ct < 16; ++ct)
#pragma unroll
      for (int j = 0; j < 8; ++j) O[ct][j] *= alpha[j];

    // --- P: D-layout -> A-layout via per-wave LDS roundtrip ------------
    bf16_t* pw = Pt[wv];
#pragma unroll
    for (int t = 0; t < 2; ++t)
#pragma unroll
      for (int j = 0; j < 8; ++j)
        pw[(j + hi * 8) * 32 + t * 16 + ln] = (bf16_t)S[t][j];
    asm volatile("s_wait_dscnt 0x0" ::: "memory");
    v16bf Pa = load_a_bf16(pw, 32, lane);

    // --- O += P x V ----------------------------------------------------
#pragma unroll
    for (int ct = 0; ct < 16; ++ct) {
      const bf16_t* p = &Vt[cur][(size_t)(ct * 16 + ln) * 32 + hi * 16];
      v8bf lo = *(const v8bf*)(p);
      v8bf hh = *(const v8bf*)(p + 8);
      v16bf bb;
#pragma unroll
      for (int e = 0; e < 8; ++e) { bb[e] = lo[e]; bb[8 + e] = hh[e]; }
      O[ct] = __builtin_amdgcn_wmma_f32_16x16x32_bf16(false, Pa, false, bb,
                                                      (short)0, O[ct], false, false);
    }
    __syncthreads();   // all reads of 'cur' done before it is re-staged
  }

  // --- finalize: divide by row sums, store bf16 [n][c] ------------------
  bf16_t* ob = o + (size_t)b * N_ * C_;
  float rl[8];
#pragma unroll
  for (int j = 0; j < 8; ++j) rl[j] = 1.0f / lrow[j];
#pragma unroll
  for (int ct = 0; ct < 16; ++ct) {
    int c = ct * 16 + ln;
#pragma unroll
    for (int j = 0; j < 8; ++j) {
      int row = qrow0 + j + hi * 8;
      ob[(size_t)row * C_ + c] = (bf16_t)(O[ct][j] * rl[j]);
    }
  }
}

// ---------------------------------------------------------------------------
// Kernel 4: output projection + bias + residual (residual = normalized x).
// out[b][o][n] = sum_c W[o][c]*attn[n][c] + bias[o] + xn[b][o][n]
// Register-blocked 32x64 per wave like the QKV GEMM.
// ---------------------------------------------------------------------------
__global__ void proj_kernel(const bf16_t* __restrict__ ao,
                            const bf16_t* __restrict__ ow_bf,
                            const float* __restrict__ obias,
                            const float* __restrict__ xn,
                            float* __restrict__ out) {
  const int wave = (blockIdx.x * blockDim.x + threadIdx.x) >> 5;  // 0..4095
  const int lane = threadIdx.x & 31;
  const int nt4 = wave & 15;            // 16 x 64-col strips
  const int t2  = wave >> 4;
  const int mt2 = t2 & 7;               // 8 x 32-row strips (256 rows)
  const int b   = t2 >> 3;

  const bf16_t* Bb = ao + (size_t)b * N_ * C_ + (size_t)(nt4 * 64) * C_;
  const bf16_t* Ab = ow_bf + (size_t)(mt2 * 32) * C_;

  v8f Acc[2][4];
#pragma unroll
  for (int i = 0; i < 2; ++i)
#pragma unroll
    for (int j = 0; j < 4; ++j) Acc[i][j] = (v8f){};

#pragma unroll
  for (int kk = 0; kk < C_; kk += 32) {
    v16bf Af[2], Bf[4];
#pragma unroll
    for (int i = 0; i < 2; ++i)
      Af[i] = load_a_bf16(Ab + (size_t)(i * 16) * C_ + kk, C_, lane);
#pragma unroll
    for (int j = 0; j < 4; ++j)
      Bf[j] = load_b_ctg(Bb + (size_t)(j * 16) * C_ + kk, C_, lane);
#pragma unroll
    for (int i = 0; i < 2; ++i)
#pragma unroll
      for (int j = 0; j < 4; ++j)
        Acc[i][j] = __builtin_amdgcn_wmma_f32_16x16x32_bf16(
            false, Af[i], false, Bf[j], (short)0, Acc[i][j], false, false);
  }

  const int hi = lane >> 4, ln = lane & 15;
#pragma unroll
  for (int i = 0; i < 2; ++i) {
    const int oc0 = mt2 * 32 + i * 16 + hi * 8;
#pragma unroll
    for (int j = 0; j < 4; ++j) {
      const int n = nt4 * 64 + j * 16 + ln;
#pragma unroll
      for (int r = 0; r < 8; ++r) {
        int oc = oc0 + r;
        size_t idx = ((size_t)b * C_ + oc) * N_ + n;
        out[idx] = Acc[i][j][r] + obias[oc] + xn[idx];
      }
    }
  }
}

// ---------------------------------------------------------------------------
extern "C" void kernel_launch(void* const* d_in, const int* in_sizes, int n_in,
                              void* d_out, int out_size, void* d_ws, size_t ws_size,
                              hipStream_t stream) {
  const float* x     = (const float*)d_in[0];
  const float* gn_w  = (const float*)d_in[1];
  const float* gn_b  = (const float*)d_in[2];
  const float* qkv_w = (const float*)d_in[3];
  const float* qkv_b = (const float*)d_in[4];
  const float* out_w = (const float*)d_in[5];
  const float* out_b = (const float*)d_in[6];
  float* out = (float*)d_out;

  char* ws = (char*)d_ws;
  size_t off = 0;
  float*  xn   = (float*)(ws + off);  off += (size_t)B_ * C_ * N_ * sizeof(float);
  bf16_t* xnT  = (bf16_t*)(ws + off); off += (size_t)B_ * N_ * C_ * sizeof(bf16_t);
  bf16_t* q    = (bf16_t*)(ws + off); off += (size_t)B_ * N_ * C_ * sizeof(bf16_t);
  bf16_t* kk   = (bf16_t*)(ws + off); off += (size_t)B_ * N_ * C_ * sizeof(bf16_t);
  bf16_t* vT   = (bf16_t*)(ws + off); off += (size_t)B_ * C_ * N_ * sizeof(bf16_t);
  bf16_t* ao   = (bf16_t*)(ws + off); off += (size_t)B_ * N_ * C_ * sizeof(bf16_t);
  bf16_t* qwbf = (bf16_t*)(ws + off); off += (size_t)OC3_ * C_ * sizeof(bf16_t);
  bf16_t* owbf = (bf16_t*)(ws + off); off += (size_t)C_ * C_ * sizeof(bf16_t);

  cvt_kernel<<<(OC3_ * C_) / 256, 256, 0, stream>>>(qkv_w, qwbf, OC3_ * C_);
  cvt_kernel<<<(C_ * C_) / 256, 256, 0, stream>>>(out_w, owbf, C_ * C_);

  gn_kernel<<<B_ * G_, 256, 0, stream>>>(x, gn_w, gn_b, xn, xnT);

  // 32 batches * 24 * 16 (32x64 wave tiles) = 12288 waves / 8 per block
  qkv_kernel<<<1536, 256, 0, stream>>>(xnT, qwbf, qkv_b, q, kk, vT);

  attn_kernel<<<B_ * 8, 256, 0, stream>>>(q, kk, vT, ao);

  // 32 batches * 8 * 16 (32x64 wave tiles) = 4096 waves / 8 per block
  proj_kernel<<<512, 256, 0, stream>>>(ao, owbf, out_b, xn, out);
}
